// NeuralLV_56427280335495
// MI455X (gfx1250) — matrix operand order repaired and checked by reading, not compiled, not run on hardware
//
#include <hip/hip_runtime.h>
#include <hip/hip_bf16.h>

typedef __attribute__((ext_vector_type(16))) _Float16 v16h;
typedef __attribute__((ext_vector_type(8)))  float    v8f;
typedef __attribute__((ext_vector_type(4)))  int      v4i;

#define N_STEPS   96
#define BATCH     16384
#define N_MAT     8
#define N_STRIKES 8
#define PERIOD    12
#define LAYER     128
#define NH        3
#define DT_F      (1.0f/96.0f)
#define SQRT_DT_F 0.10206207262f
#define RFR_F     0.03f
#define DF_F      0.99968754883f   /* exp(-0.03/96) */
#define S0_F      1.0f

#define PATHS_PER_BLOCK 64
#define THREADS         256
#define HSTRIDE         136        /* padded f16 row stride for activations */

/* workspace layout (in halves) */
#define LEVWH_OFF 0
#define HEDWH_OFF 393216
#define HEDWO_OFF 786432
#define LEVWO_OFF 851968
#define TOTAL_HALVES 852992
#define SUMS_BYTE_OFF (TOTAL_HALVES*2)   /* float sums[64], sumsq[64] after halves */

/* ---- CDNA5 async global->LDS staging (ASYNCcnt path), with safe fallbacks ---- */
#if defined(__has_builtin)
#  if __has_builtin(__builtin_amdgcn_global_load_async_to_lds_b128)
#    define USE_ASYNC_LDS 1
#  endif
#endif

typedef __attribute__((address_space(1))) v4i* g4p;   /* global int4*  */
typedef __attribute__((address_space(3))) v4i* l4p;   /* LDS int4*     */

union V16Q  { v16h v; uint4 q[2]; };
union F8    { v8f  v; float f[8]; };

static __device__ inline void wait_async_lds() {
#if defined(USE_ASYNC_LDS)
#  if __has_builtin(__builtin_amdgcn_s_wait_asynccnt)
  __builtin_amdgcn_s_wait_asynccnt(0);
#  else
  asm volatile("s_wait_asynccnt 0x0" ::: "memory");
#  endif
#endif
}

/* cooperative stage of nhalf f16 weights global->LDS */
static __device__ inline void stageW(_Float16* dst, const _Float16* src, int nhalf, int tid) {
  const int n16 = nhalf >> 3;
#if defined(USE_ASYNC_LDS)
  g4p s = (g4p)(_Float16*)src;
  l4p d = (l4p)dst;
  for (int i = tid; i < n16; i += THREADS)
    __builtin_amdgcn_global_load_async_to_lds_b128(s + i, d + i, 0, 0);
#else
  const uint4* s = (const uint4*)src;
  uint4*       d = (uint4*)dst;
  for (int i = tid; i < n16; i += THREADS) d[i] = s[i];
#endif
}

/* ---- A fragment (16x32 f16, M in lanes; ISA 7.12.2): two b128 LDS loads ---- */
static __device__ inline v16h loadA_lds(const _Float16* W, int m_base, int kbase,
                                        int lane, int ldw) {
  V16Q r;
  const int m   = m_base + (lane & 15);
  const int klo = (lane & 16) ? 8 : 0;
  const _Float16* base = W + m * ldw + kbase + klo;   /* 16B aligned */
  r.q[0] = *(const uint4*)(base);        /* K = klo+0..7      */
  r.q[1] = *(const uint4*)(base + 16);   /* K = 16+klo+0..7   */
  return r.v;
}

/* ---- B fragment (32x16 f16, N in lanes): K = klo+0..7 and 16+klo+0..7, two b128 ---- */
static __device__ inline v16h loadB_lds(const _Float16* H, int n_base, int kbase, int lane) {
  V16Q r;
  const int p   = n_base + (lane & 15);
  const int klo = (lane & 16) ? 8 : 0;
  const _Float16* base = H + p * HSTRIDE + kbase + klo;  /* 16B aligned */
  r.q[0] = *(const uint4*)(base);        /* K = klo+0..7      */
  r.q[1] = *(const uint4*)(base + 16);   /* K = 16+klo+0..7   */
  return r.v;
}

static __device__ inline float softplusf(float x) {
  return (x > 20.0f) ? x : log1pf(__expf(x));
}

/* one 128x128 @ 128x64 hidden layer: WMMA + bias + leaky_relu, write f16 to hd */
static __device__ inline void hiddenLayer(const _Float16* Wb, const _Float16* hs,
                                          _Float16* hd, const float* bias,
                                          int wid, int lane) {
  const v8f vzero = {0.f,0.f,0.f,0.f,0.f,0.f,0.f,0.f};
  v8f acc[4] = {vzero, vzero, vzero, vzero};
  const int m_base = wid * 16;
#pragma unroll
  for (int kt = 0; kt < 4; ++kt) {
    const v16h a = loadA_lds(Wb, m_base, kt * 32, lane, 128);
#pragma unroll
    for (int nt = 0; nt < 4; ++nt) {
      const v16h b = loadB_lds(hs, nt * 16, kt * 32, lane);
      acc[nt] = __builtin_amdgcn_wmma_f32_16x16x32_f16(false, a, false, b,
                                                       (short)0, acc[nt], false, false);
    }
  }
  const int mb = m_base + ((lane & 16) ? 8 : 0);
#pragma unroll
  for (int nt = 0; nt < 4; ++nt) {
    const int p = nt * 16 + (lane & 15);
    F8 c; c.v = acc[nt];
    _Float16 out[8];
#pragma unroll
    for (int r = 0; r < 8; ++r) {
      float x = c.f[r] + bias[mb + r];
      x = (x > 0.f) ? x : 0.01f * x;
      out[r] = (_Float16)x;
    }
    *(uint4*)(hd + p * HSTRIDE + mb) = *(uint4*)out;
  }
}

__global__ void prep_kernel(const float* levWh, const float* hedWh,
                            const float* hedWo, const float* levWo,
                            _Float16* dst, float* sums) {
  const int gid = blockIdx.x * blockDim.x + threadIdx.x;
  const int str = gridDim.x * blockDim.x;
  for (int i = gid; i < TOTAL_HALVES; i += str) {
    float v;
    if      (i < HEDWH_OFF) v = levWh[i];
    else if (i < HEDWO_OFF) v = hedWh[i - HEDWH_OFF];
    else if (i < LEVWO_OFF) v = hedWo[i - HEDWO_OFF];
    else                    v = levWo[i - LEVWO_OFF];
    dst[i] = (_Float16)v;
  }
  if (gid < 128) sums[gid] = 0.0f;   /* sums[64] + sumsq[64] */
}

__global__ void __launch_bounds__(THREADS)
neurallv_main(const float* __restrict__ normals, const float* __restrict__ strikes,
              const float* __restrict__ lev_Win, const float* __restrict__ lev_bin,
              const float* __restrict__ lev_bh,  const float* __restrict__ lev_WoutF,
              const float* __restrict__ lev_bout,
              const float* __restrict__ hed_Win, const float* __restrict__ hed_bin,
              const float* __restrict__ hed_bh,  const float* __restrict__ hed_bout,
              const _Float16* __restrict__ wsH,
              float* __restrict__ sums, float* __restrict__ sumsq) {
  extern __shared__ char smem[];
  _Float16* Wb    = (_Float16*)smem;     /* 16384 halves */
  _Float16* hA    = Wb + 16384;          /*  8704 halves */
  _Float16* hB    = hA + 8704;           /*  8704 halves */
  float*    Sprev = (float*)(hB + 8704); /* 64 */
  float*    Snew  = Sprev + 64;          /* 64 */
  float*    scA   = Snew + 64;           /* 64 */

  const int tid  = threadIdx.x;
  const int wid  = tid >> 5;
  const int lane = tid & 31;
  const int pbase = blockIdx.x * PATHS_PER_BLOCK;

  const _Float16* levWh_h = wsH + LEVWH_OFF;
  const _Float16* hedWh_h = wsH + HEDWH_OFF;
  const _Float16* hedWo_h = wsH + HEDWO_OFF;

  /* persistent hedging accumulator: this wave's 16 rows x 32 paths */
  const v8f vzero = {0.f,0.f,0.f,0.f,0.f,0.f,0.f,0.f};
  v8f hacc[2] = {vzero, vzero};

  if (tid < PATHS_PER_BLOCK) Sprev[tid] = S0_F;
  __syncthreads();

  float dfp = 1.0f;  /* df^(i-1) */

  for (int i = 1; i <= N_STEPS; ++i) {
    const int   idxm = (i - 1) / PERIOD;
    const float tcur = (float)(i - 1) * DT_F;

    /* ---------- leverage MLP: input layer (K=2, plain VALU) -> hA ---------- */
    {
      const float* Wi = lev_Win + idxm * LAYER * 2;
      const float* bi = lev_bin + idxm * LAYER;
      for (int e = tid; e < LAYER * PATHS_PER_BLOCK; e += THREADS) {
        const int r = e & 127, p = e >> 7;
        float x = Wi[r * 2 + 0] * tcur + Wi[r * 2 + 1] * Sprev[p] + bi[r];
        x = (x > 0.f) ? x : 0.01f * x;
        hA[p * HSTRIDE + r] = (_Float16)x;
      }
    }
    _Float16* hs = hA;
    _Float16* hd = hB;
    for (int l = 0; l < NH; ++l) {
      __syncthreads();
      stageW(Wb, levWh_h + (size_t)(idxm * NH + l) * 16384, 16384, tid);
      wait_async_lds();
      __syncthreads();
      hiddenLayer(Wb, hs, hd, lev_bh + (idxm * NH + l) * LAYER, wid, lane);
      _Float16* t = hs; hs = hd; hd = t;
    }
    __syncthreads();

    /* ---------- lev output (1x128) + SDE path update, threads 0..63 ---------- */
    if (tid < PATHS_PER_BLOCK) {
      const int p = tid;
      float a = lev_bout[idxm];
      const _Float16* hr = hs + p * HSTRIDE;
      const float* wo = lev_WoutF + idxm * LAYER;
#pragma unroll 8
      for (int k = 0; k < LAYER; ++k) a += wo[k] * (float)hr[k];
      const float lv    = softplusf(a);
      const float S     = Sprev[p];
      const float drift = S * RFR_F / (1.f + fabsf(S * RFR_F) * SQRT_DT_F);
      const float diff  = S * lv    / (1.f + fabsf(S * lv)    * SQRT_DT_F);
      const float dW    = SQRT_DT_F * normals[(size_t)(i - 1) * BATCH + pbase + p];
      Snew[p] = S + drift * DT_F + diff * dW;
      scA[p]  = dfp * S * diff * dW;   /* df^(i-1) * (S_prev*diffusion) * dW */
    }

    /* ---------- hedge MLP: input layer -> hA (lev result lives in hB) ---------- */
    {
      const float* Wi = hed_Win + idxm * LAYER * 2;
      const float* bi = hed_bin + idxm * LAYER;
      for (int e = tid; e < LAYER * PATHS_PER_BLOCK; e += THREADS) {
        const int r = e & 127, p = e >> 7;
        float x = Wi[r * 2 + 0] * tcur + Wi[r * 2 + 1] * Sprev[p] + bi[r];
        x = (x > 0.f) ? x : 0.01f * x;
        hA[p * HSTRIDE + r] = (_Float16)x;
      }
    }
    hs = hA; hd = hB;
    for (int l = 0; l < NH; ++l) {
      __syncthreads();
      stageW(Wb, hedWh_h + (size_t)(idxm * NH + l) * 16384, 16384, tid);
      wait_async_lds();
      __syncthreads();
      hiddenLayer(Wb, hs, hd, hed_bh + (idxm * NH + l) * LAYER, wid, lane);
      _Float16* t = hs; hs = hd; hd = t;
    }

    /* ---------- hedge output (64x128 WMMA) + hedging accumulation ---------- */
    __syncthreads();
    stageW(Wb, hedWo_h + (size_t)idxm * 8192, 8192, tid);
    wait_async_lds();
    __syncthreads();
    {
      const int m_base = (wid & 3) * 16;
      const int ntb    = (wid >> 2) * 2;
      v8f acc[2] = {vzero, vzero};
#pragma unroll
      for (int kt = 0; kt < 4; ++kt) {
        const v16h a = loadA_lds(Wb, m_base, kt * 32, lane, 128);
#pragma unroll
        for (int j = 0; j < 2; ++j) {
          const v16h b = loadB_lds(hs, (ntb + j) * 16, kt * 32, lane);
          acc[j] = __builtin_amdgcn_wmma_f32_16x16x32_f16(false, a, false, b,
                                                          (short)0, acc[j], false, false);
        }
      }
      const int mb = m_base + ((lane & 16) ? 8 : 0);
#pragma unroll
      for (int j = 0; j < 2; ++j) {
        const int   p = (ntb + j) * 16 + (lane & 15);
        const float s = scA[p];
        F8 c; c.v = acc[j];
        F8 h; h.v = hacc[j];
#pragma unroll
        for (int r = 0; r < 8; ++r)
          h.f[r] += s * (c.f[r] + hed_bout[idxm * 64 + mb + r]);
        hacc[j] = h.v;
      }
    }

    /* ---------- maturity: discounted payoff - hedging, sum & sumsq ---------- */
    if ((i % PERIOD) == 0) {
      const int   mat = i / PERIOD - 1;
      const float dfi = dfp * DF_F;   /* df^i */
      const int   ntb  = (wid >> 2) * 2;
      const int   sidx = 2 * (wid & 3) + ((lane & 16) ? 1 : 0);
      const float K = strikes[sidx];
      float sum = 0.f, sq = 0.f;
#pragma unroll
      for (int j = 0; j < 2; ++j) {
        const int p = (ntb + j) * 16 + (lane & 15);
        float pay = Snew[p] - K;
        pay = (pay > 0.f) ? pay : 0.f;
        F8 h; h.v = hacc[j];
        float hv = 0.f;
#pragma unroll
        for (int r = 0; r < 8; ++r) hv = (r == mat) ? h.f[r] : hv;
        const float sim = dfi * pay - hv;
        sum += sim; sq += sim * sim;
      }
#pragma unroll
      for (int off = 1; off < 16; off <<= 1) {
        sum += __shfl_xor(sum, off, 32);
        sq  += __shfl_xor(sq,  off, 32);
      }
      if ((lane & 15) == 0) {
        atomicAdd(&sums [sidx * N_MAT + mat], sum);
        atomicAdd(&sumsq[sidx * N_MAT + mat], sq);
      }
    }

    __syncthreads();
    if (tid < PATHS_PER_BLOCK) Sprev[tid] = Snew[tid];
    dfp *= DF_F;
    __syncthreads();
  }
}

__global__ void fin_kernel(const float* sums, const float* sumsq, float* out) {
  const int t = threadIdx.x;
  if (t < 64) {
    const float s = sums[t], q = sumsq[t];
    out[t]      = s / (float)BATCH;
    out[64 + t] = (q - s * s / (float)BATCH) / (float)(BATCH - 1);
  }
}

extern "C" void kernel_launch(void* const* d_in, const int* in_sizes, int n_in,
                              void* d_out, int out_size, void* d_ws, size_t ws_size,
                              hipStream_t stream) {
  (void)in_sizes; (void)n_in; (void)out_size; (void)ws_size;
  const float* normals  = (const float*)d_in[0];
  const float* strikes  = (const float*)d_in[1];
  const float* lev_Win  = (const float*)d_in[2];
  const float* lev_bin  = (const float*)d_in[3];
  const float* lev_Wh   = (const float*)d_in[4];
  const float* lev_bh   = (const float*)d_in[5];
  const float* lev_Wout = (const float*)d_in[6];
  const float* lev_bout = (const float*)d_in[7];
  const float* hed_Win  = (const float*)d_in[8];
  const float* hed_bin  = (const float*)d_in[9];
  const float* hed_Wh   = (const float*)d_in[10];
  const float* hed_bh   = (const float*)d_in[11];
  const float* hed_Wout = (const float*)d_in[12];
  const float* hed_bout = (const float*)d_in[13];

  _Float16* wsH   = (_Float16*)d_ws;
  float*    sums  = (float*)((char*)d_ws + SUMS_BYTE_OFF);
  float*    sumsq = sums + 64;

  prep_kernel<<<512, 256, 0, stream>>>(lev_Wh, hed_Wh, hed_Wout, lev_Wout, wsH, sums);

  const size_t ldsBytes = (size_t)(16384 + 8704 + 8704) * 2 + 3 * 64 * 4;
  neurallv_main<<<BATCH / PATHS_PER_BLOCK, THREADS, ldsBytes, stream>>>(
      normals, strikes, lev_Win, lev_bin, lev_bh, lev_Wout, lev_bout,
      hed_Win, hed_bin, hed_bh, hed_bout, wsH, sums, sumsq);

  fin_kernel<<<1, 64, 0, stream>>>(sums, sumsq, (float*)d_out);
}
